// TriangularSelfAttention_12618613916184
// MI455X (gfx1250) — matrix operand order repaired and checked by reading, not compiled, not run on hardware
//
#include <hip/hip_runtime.h>
#include <hip/hip_bf16.h>
#include <math.h>

typedef __attribute__((ext_vector_type(16))) _Float16 v16h;
typedef __attribute__((ext_vector_type(8)))  float    v8f;
typedef int v4i_vec __attribute__((vector_size(16)));   // matches builtin param

constexpr int kN  = 256;   // sequence length
constexpr int kC  = 128;   // channels
constexpr int kH  = 8;     // heads
constexpr int kD  = 16;    // head dim
constexpr int kNN = kN * kN;

// ---------------- async global->LDS staging (gfx1250 ASYNCcnt path) --------

__device__ __forceinline__ void async_copy_b128(void* lds_dst, const void* gsrc) {
#if defined(__AMDGCN__) && __has_builtin(__builtin_amdgcn_global_load_async_to_lds_b128)
  __builtin_amdgcn_global_load_async_to_lds_b128(
      (__attribute__((address_space(1))) v4i_vec*)gsrc,
      (__attribute__((address_space(3))) v4i_vec*)lds_dst, 0, 0);
#else
  *(float4*)lds_dst = *(const float4*)gsrc;
#endif
}

__device__ __forceinline__ void async_join() {
#if defined(__AMDGCN__) && __has_builtin(__builtin_amdgcn_global_load_async_to_lds_b128)
#if __has_builtin(__builtin_amdgcn_s_wait_asynccnt)
  __builtin_amdgcn_s_wait_asynccnt(0);
#else
  asm volatile("s_wait_asynccnt 0x0" ::: "memory");
#endif
#endif
  __syncthreads();
}

// ---------------- WMMA fragment helpers (wave32, gfx1250 layouts) ----------

__device__ __forceinline__ v8f wmma16(v16h a, v16h b, v8f c) {
  return __builtin_amdgcn_wmma_f32_16x16x32_f16(
      /*neg_a=*/false, a, /*neg_b=*/false, b,
      /*c_mod=*/(short)0, c, /*reuse_a=*/false, /*reuse_b=*/false);
}

// A (16x32 f16, M x K): lanes 0-15 -> M=lane, K=0..15; lanes 16-31 ->
// M=lane-16, K=16..31. element (m,k) at base[m*ld + k].
__device__ __forceinline__ v16h frag_a_f16(const _Float16* base, int ld) {
  const int lane = threadIdx.x & 31;
  const int m = lane & 15, hi = lane >> 4;
  v16h a;
#pragma unroll
  for (int r = 0; r < 8; ++r) {
    const int kk = hi * 16 + 2 * r;
    a[2 * r]     = base[m * ld + kk];
    a[2 * r + 1] = base[m * ld + kk + 1];
  }
  return a;
}

// Same, but only K=0..15 valid (K=16..31 zero-padded). Used for head dim 16.
__device__ __forceinline__ v16h frag_a_f16_k16(const _Float16* base, int ld) {
  const int lane = threadIdx.x & 31;
  const int m = lane & 15, hi = lane >> 4;
  v16h a;
#pragma unroll
  for (int r = 0; r < 8; ++r) {
    const int kk = hi * 16 + 2 * r;
    a[2 * r]     = (kk < 16)     ? base[m * ld + kk]     : (_Float16)0.0f;
    a[2 * r + 1] = (kk + 1 < 16) ? base[m * ld + kk + 1] : (_Float16)0.0f;
  }
  return a;
}

// B (32x16 f16, K x N): lanes 0-15 -> N=lane, K=0..15; lanes 16-31 -> K=16..31.
// Row-major source: element (k,n) at base[k*ld + n].
__device__ __forceinline__ v16h frag_b_row(const _Float16* base, int ld) {
  const int lane = threadIdx.x & 31;
  const int n = lane & 15;
  const int kb = (lane >> 4) << 4;
  v16h b;
#pragma unroll
  for (int r = 0; r < 8; ++r) {
    const int kk = kb + 2 * r;
    b[2 * r]     = base[kk * ld + n];
    b[2 * r + 1] = base[(kk + 1) * ld + n];
  }
  return b;
}

// Column-major source: element (k,n) at base[n*ld + k].
__device__ __forceinline__ v16h frag_b_col(const _Float16* base, int ld) {
  const int lane = threadIdx.x & 31;
  const int n = lane & 15;
  const int kb = (lane >> 4) << 4;
  v16h b;
#pragma unroll
  for (int r = 0; r < 8; ++r) {
    const int kk = kb + 2 * r;
    b[2 * r]     = base[n * ld + kk];
    b[2 * r + 1] = base[n * ld + kk + 1];
  }
  return b;
}

// Column-major, only K=0..15 valid (zero-pad upper half).
__device__ __forceinline__ v16h frag_b_col_k16(const _Float16* base, int ld) {
  const int lane = threadIdx.x & 31;
  const int n = lane & 15;
  const int kb = (lane >> 4) << 4;
  v16h b;
#pragma unroll
  for (int r = 0; r < 8; ++r) {
    const int kk = kb + 2 * r;
    b[2 * r]     = (kk < 16)     ? base[n * ld + kk]     : (_Float16)0.0f;
    b[2 * r + 1] = (kk + 1 < 16) ? base[n * ld + kk + 1] : (_Float16)0.0f;
  }
  return b;
}

// ---------------- Kernel 0: one-shot weight f32 -> f16 convert -------------
// 5 matrices of 128x128, laid out consecutively in ws.
__global__ __launch_bounds__(256) void wcvt_kernel(
    const float* __restrict__ w0, const float* __restrict__ w1,
    const float* __restrict__ w2, const float* __restrict__ w3,
    const float* __restrict__ w4, _Float16* __restrict__ dst) {
  const int t = blockIdx.x * 256 + threadIdx.x;  // 0 .. 5*16384-1
  const int mat = t >> 14;
  const int idx = t & 16383;
  const float* srcs[5] = {w0, w1, w2, w3, w4};
  dst[t] = (_Float16)srcs[mat][idx];
}

// ---------------- Kernel 1: LayerNorm -> f16 normed ------------------------
__global__ __launch_bounds__(256) void ln_kernel(
    const float* __restrict__ x, const float* __restrict__ gamma,
    const float* __restrict__ beta, _Float16* __restrict__ xn) {
  const int wave = threadIdx.x >> 5;
  const int lane = threadIdx.x & 31;
  const int p = blockIdx.x * 8 + wave;
  const float4 v = reinterpret_cast<const float4*>(x + (size_t)p * kC)[lane];
  float s = v.x + v.y + v.z + v.w;
#pragma unroll
  for (int o = 16; o > 0; o >>= 1) s += __shfl_xor(s, o, 32);
  const float mu = s * (1.0f / kC);
  const float d0 = v.x - mu, d1 = v.y - mu, d2 = v.z - mu, d3 = v.w - mu;
  float vs = d0 * d0 + d1 * d1 + d2 * d2 + d3 * d3;
#pragma unroll
  for (int o = 16; o > 0; o >>= 1) vs += __shfl_xor(vs, o, 32);
  const float rstd = rsqrtf(vs * (1.0f / kC) + 1e-5f);
  const int c0 = lane * 4;
  _Float16* dst = xn + (size_t)p * kC + c0;
  dst[0] = (_Float16)(d0 * rstd * gamma[c0 + 0] + beta[c0 + 0]);
  dst[1] = (_Float16)(d1 * rstd * gamma[c0 + 1] + beta[c0 + 1]);
  dst[2] = (_Float16)(d2 * rstd * gamma[c0 + 2] + beta[c0 + 2]);
  dst[3] = (_Float16)(d3 * rstd * gamma[c0 + 3] + beta[c0 + 3]);
}

// ---------------- Kernel 2: q/k/v projections (WMMA) -----------------------
// grid = (kNN/64, 3). Block computes 64(M) x 128(N) for one weight.
// 8 waves = 8 N-tiles; each wave holds 4 M-subtile accumulators and reuses
// each B fragment across 4 chained WMMAs.
__global__ __launch_bounds__(256) void qkv_kernel(
    const _Float16* __restrict__ xn, const _Float16* __restrict__ wf,
    _Float16* __restrict__ qo, _Float16* __restrict__ ko,
    _Float16* __restrict__ vo) {
  __shared__ _Float16 sW[kC * kC];   // weight [o][c], f16 (32 KB)
  __shared__ _Float16 sA[64 * kC];   // normed tile [m][c]   (16 KB)
  const int tid = threadIdx.x;
  const int which = blockIdx.y;
  const int m0 = blockIdx.x * 64;
  const _Float16* Wsrc = wf + (size_t)which * (kC * kC);
  for (int t = tid; t < (kC * kC) / 8; t += 256)
    async_copy_b128(sW + t * 8, Wsrc + t * 8);
  for (int t = tid; t < (64 * kC) / 8; t += 256)
    async_copy_b128(sA + t * 8, xn + (size_t)m0 * kC + t * 8);
  async_join();

  const int wid = tid >> 5;  // N-tile
  v8f acc[4];
#pragma unroll
  for (int ms = 0; ms < 4; ++ms)
    acc[ms] = v8f{0.f, 0.f, 0.f, 0.f, 0.f, 0.f, 0.f, 0.f};
#pragma unroll
  for (int ks = 0; ks < 4; ++ks) {
    const v16h b = frag_b_col(sW + wid * 16 * kC + ks * 32, kC);
#pragma unroll
    for (int ms = 0; ms < 4; ++ms) {
      const v16h a = frag_a_f16(sA + ms * 16 * kC + ks * 32, kC);
      acc[ms] = wmma16(a, b, acc[ms]);
    }
  }
  const float scl = (which == 0) ? 0.25f : 1.0f;  // fold 1/sqrt(D) into q
  _Float16* dst = (which == 0) ? qo : ((which == 1) ? ko : vo);
  const int lane = tid & 31;
  const int n = lane & 15;
  const int mh = (lane >> 4) * 8;
#pragma unroll
  for (int ms = 0; ms < 4; ++ms)
#pragma unroll
    for (int r = 0; r < 8; ++r)
      dst[(size_t)(m0 + ms * 16 + mh + r) * kC + wid * 16 + n] =
          (_Float16)(acc[ms][r] * scl);
}

// ---------------- Kernel 3: per-row attention (WMMA + softmax) -------------
// grid = (16 q-tiles, 8 heads, 256 rows). Block: 16 queries vs all 256 keys.
__global__ __launch_bounds__(256) void attn_kernel(
    const _Float16* __restrict__ qb, const _Float16* __restrict__ kbuf,
    const _Float16* __restrict__ vbuf, const unsigned char* __restrict__ mask,
    _Float16* __restrict__ ao) {
  __shared__ _Float16 sK[kN * kD];   // [key][d]  8 KB
  __shared__ _Float16 sV[kN * kD];   // [key][d]  8 KB
  __shared__ _Float16 sQ[16 * kD];   // [q][d]
  __shared__ float    sS[16 * kN];   // raw scores 16 KB
  __shared__ _Float16 sP[16 * kN];   // softmax probs 8 KB
  const int tid = threadIdx.x;
  const int qt = blockIdx.x;
  const int h  = blockIdx.y;
  const int i  = blockIdx.z;
  const size_t rowbase = (size_t)i * kN;

  for (int t = tid; t < kN * 2; t += 256) {      // 512 x 16B chunks each
    const int key = t >> 1, part = (t & 1) * 8;
    const size_t g = (rowbase + key) * kC + h * kD + part;
    async_copy_b128(sK + key * kD + part, kbuf + g);
    async_copy_b128(sV + key * kD + part, vbuf + g);
  }
  if (tid < 32) {
    const int qq = tid >> 1, part = (tid & 1) * 8;
    async_copy_b128(sQ + qq * kD + part,
                    qb + (rowbase + qt * 16 + qq) * kC + h * kD + part);
  }
  async_join();

  // scores: 16x256, K-dim = 16 (zero-padded to 32); 16 N-tiles over 8 waves
  const int wid = tid >> 5;
  const int lane = tid & 31;
  for (int nt = wid; nt < 16; nt += 8) {
    const v16h a = frag_a_f16_k16(sQ, kD);
    const v16h b = frag_b_col_k16(sK + nt * 16 * kD, kD);  // B[k=d][n=key]
    v8f acc = {0.f, 0.f, 0.f, 0.f, 0.f, 0.f, 0.f, 0.f};
    acc = wmma16(a, b, acc);
    const int n = lane & 15;
    const int mh = (lane >> 4) * 8;
#pragma unroll
    for (int r = 0; r < 8; ++r) sS[(mh + r) * kN + nt * 16 + n] = acc[r];
  }
  __syncthreads();

  // masked softmax: 16 threads per query row, shfl reductions in 16-lane groups
  {
    const int r = tid >> 4;
    const int sub = tid & 15;
    const bool rowv = (mask[i] != 0);
    float vals[16];
    float mx = -3.0e38f;
#pragma unroll
    for (int j = 0; j < 16; ++j) {
      const int kk = sub * 16 + j;
      float s = sS[r * kN + kk];
      s = (rowv && (mask[kk] != 0)) ? s : -1.0e9f;
      vals[j] = s;
      mx = fmaxf(mx, s);
    }
#pragma unroll
    for (int o = 8; o > 0; o >>= 1) mx = fmaxf(mx, __shfl_xor(mx, o, 32));
    float sum = 0.0f;
#pragma unroll
    for (int j = 0; j < 16; ++j) {
      vals[j] = __expf(vals[j] - mx);
      sum += vals[j];
    }
#pragma unroll
    for (int o = 8; o > 0; o >>= 1) sum += __shfl_xor(sum, o, 32);
    const float inv = 1.0f / sum;
#pragma unroll
    for (int j = 0; j < 16; ++j)
      sP[r * kN + sub * 16 + j] = (_Float16)(vals[j] * inv);
  }
  __syncthreads();

  // attn @ V : (16x256)(256x16), 8 K-steps of 32, wave 0
  if (wid == 0) {
    v8f acc = {0.f, 0.f, 0.f, 0.f, 0.f, 0.f, 0.f, 0.f};
#pragma unroll
    for (int ks = 0; ks < 8; ++ks) {
      const v16h a = frag_a_f16(sP + ks * 32, kN);
      const v16h b = frag_b_row(sV + ks * 32 * kD, kD);  // B[k=key][n=d]
      acc = wmma16(a, b, acc);
    }
    const int n = lane & 15;
    const int mh = (lane >> 4) * 8;
#pragma unroll
    for (int r = 0; r < 8; ++r)
      ao[(rowbase + qt * 16 + mh + r) * kC + h * kD + n] = (_Float16)acc[r];
  }
}

// ---------------- Kernel 4: out-proj + gate + blend ------------------------
// grid = (kNN/16, 2 N-halves). Waves 0-3: out GEMM tiles; waves 4-7: gate.
__global__ __launch_bounds__(256) void out_kernel(
    const _Float16* __restrict__ ao, const float* __restrict__ pair,
    const _Float16* __restrict__ wo16, const float* __restrict__ bo,
    const _Float16* __restrict__ wg16, const float* __restrict__ bg,
    const _Float16* __restrict__ xn, float* __restrict__ out) {
  __shared__ _Float16 sWo[64 * kC];   // [n][c] 16 KB
  __shared__ _Float16 sWg[64 * kC];   // 16 KB
  __shared__ _Float16 sAo[16 * kC];   // 4 KB
  __shared__ _Float16 sAp[16 * kC];   // 4 KB
  __shared__ float sOut[16 * 64];
  __shared__ float sGate[16 * 64];
  const int tid = threadIdx.x;
  const int m0 = blockIdx.x * 16;
  const int o0 = blockIdx.y * 64;

  for (int t = tid; t < (64 * kC) / 8; t += 256) {   // 1024 chunks each
    async_copy_b128(sWo + t * 8, wo16 + (size_t)o0 * kC + t * 8);
    async_copy_b128(sWg + t * 8, wg16 + (size_t)o0 * kC + t * 8);
  }
  for (int t = tid; t < (16 * kC) / 8; t += 256)
    async_copy_b128(sAo + t * 8, ao + (size_t)m0 * kC + t * 8);
  const bool more = (blockIdx.x + 1) < (kNN / 16);
  for (int t = tid; t < 16 * kC; t += 256) {
    const size_t g = (size_t)m0 * kC + t;
    if (more) __builtin_prefetch(&pair[g + 16 * kC], 0, 1);  // global_prefetch_b8
    sAp[t] = (_Float16)pair[g];
  }
  async_join();

  const int wid = tid >> 5;
  const int isg = wid >> 2;   // 0: out projection, 1: gate projection
  const int nt = wid & 3;
  const _Float16* A  = isg ? sAp : sAo;
  const _Float16* Wl = isg ? sWg : sWo;
  v8f acc = {0.f, 0.f, 0.f, 0.f, 0.f, 0.f, 0.f, 0.f};
#pragma unroll
  for (int ks = 0; ks < 4; ++ks) {
    const v16h a = frag_a_f16(A + ks * 32, kC);
    const v16h b = frag_b_col(Wl + nt * 16 * kC + ks * 32, kC);
    acc = wmma16(a, b, acc);
  }
  const float* bias = isg ? bg : bo;
  float* dstS = isg ? sGate : sOut;
  const int lane = tid & 31;
  const int n = lane & 15;
  const int mh = (lane >> 4) * 8;
#pragma unroll
  for (int r = 0; r < 8; ++r)
    dstS[(mh + r) * 64 + nt * 16 + n] = acc[r] + bias[o0 + nt * 16 + n];
  __syncthreads();

  for (int t = tid; t < 16 * 64; t += 256) {
    const int m = t >> 6, c = t & 63;
    const size_t g = (size_t)(m0 + m) * kC + (o0 + c);
    const float gt = 1.0f / (1.0f + __expf(-sGate[t]));
    out[g] = gt * sOut[t] + (1.0f - gt) * (float)xn[g];
  }
}

// ---------------- launch ---------------------------------------------------
extern "C" void kernel_launch(void* const* d_in, const int* in_sizes, int n_in,
                              void* d_out, int out_size, void* d_ws, size_t ws_size,
                              hipStream_t stream) {
  const float* pair         = (const float*)d_in[0];
  const unsigned char* mask = (const unsigned char*)d_in[1];  // bool mask
  const float* gamma        = (const float*)d_in[2];
  const float* beta         = (const float*)d_in[3];
  const float* wq           = (const float*)d_in[4];
  const float* wk           = (const float*)d_in[5];
  const float* wv           = (const float*)d_in[6];
  const float* wo           = (const float*)d_in[7];
  const float* bo           = (const float*)d_in[8];
  const float* wg           = (const float*)d_in[9];
  const float* bg           = (const float*)d_in[10];
  float* out = (float*)d_out;

  char* ws = (char*)d_ws;
  const size_t seg = (size_t)kNN * kC * sizeof(_Float16);  // 16 MB each
  _Float16* xn = (_Float16*)(ws + 0 * seg);
  _Float16* q  = (_Float16*)(ws + 1 * seg);
  _Float16* k  = (_Float16*)(ws + 2 * seg);
  _Float16* v  = (_Float16*)(ws + 3 * seg);
  _Float16* ao = (_Float16*)(ws + 4 * seg);
  _Float16* wf = (_Float16*)(ws + 5 * seg);  // 5 x 128x128 f16 weights
  (void)in_sizes; (void)n_in; (void)out_size; (void)ws_size;

  wcvt_kernel<<<dim3(5 * kC * kC / 256), dim3(256), 0, stream>>>(
      wq, wk, wv, wo, wg, wf);
  ln_kernel<<<dim3(kNN / 8), dim3(256), 0, stream>>>(pair, gamma, beta, xn);
  qkv_kernel<<<dim3(kNN / 64, 3), dim3(256), 0, stream>>>(xn, wf, q, k, v);
  attn_kernel<<<dim3(kN / 16, kH, kN), dim3(256), 0, stream>>>(q, k, v, mask, ao);
  out_kernel<<<dim3(kNN / 16, 2), dim3(256), 0, stream>>>(
      ao, pair, wf + 3 * kC * kC, bo, wf + 4 * kC * kC, bg, xn, out);
}
